// Model_17471926960893
// MI455X (gfx1250) — matrix operand order repaired and checked by reading, not compiled, not run on hardware
//
#include <hip/hip_runtime.h>
#include <hip/hip_bf16.h>

// ---------------- problem constants ----------------
#define B_    8
#define SEQ_  1024
#define PRED_ 1024
#define CIN_  21
#define MARK_ 4
#define D_    512
#define H_    8
#define DFF_  2048
#define L_    3
#define BUCKET_ 32
#define NHASH_  4
#define T_    2048              // SEQ + PRED
#define NB_   64                // T / BUCKET
#define DH_   64                // D / H
#define NTOK_ (B_ * T_)         // 16384
#define BH_   (B_ * H_)         // 64
#define NS_   (NHASH_ * T_)     // 8192 sorted positions per bh
#define NCHUNK_ (NS_ / BUCKET_) // 256 chunks per bh
#define SELF_ATTN_VALUE (-5e4f)
#define BK_   64                // GEMM K-chunk staged through LDS by the TDM

typedef __attribute__((ext_vector_type(16))) _Float16 v16h;
typedef __attribute__((ext_vector_type(8)))  _Float16 v8h;
typedef __attribute__((ext_vector_type(8)))  float    v8f;
typedef __attribute__((ext_vector_type(4)))  unsigned int u32x4;
typedef __attribute__((ext_vector_type(8)))  int      i32x8;
typedef __attribute__((ext_vector_type(4)))  int      i32x4;

// ---------------- WMMA fragment helpers (contiguous K order, same for A and B) ----------------
// A-matrix 16x32 f16 (MxK): lanes 0-15 -> M=lane, K base 0; lanes 16-31 -> M=lane-16, K base 16 (K contiguous).
__device__ __forceinline__ v16h load_a_frag(const _Float16* __restrict__ A, int lda,
                                            int row0, int k0) {
  const int lane = threadIdx.x & 31;
  const int m  = row0 + (lane & 15);
  const int kb = k0 + ((lane >> 4) << 4);
  const _Float16* p = A + (size_t)m * lda + kb;
  v16h a;
  *(v8h*)&a       = *(const v8h*)(p);       // K = kb .. kb+7
  *((v8h*)&a + 1) = *(const v8h*)(p + 8);   // K = kb+8 .. kb+15
  return a;
}

// B-matrix 32x16 f16 (KxN) for C = A @ W^T with W row-major (N rows, K cols):
// B[k][n] = W[n][k]; lane n = n0 + (lane&15); K contiguous: kb = k0 + 16*(lane>=16).
__device__ __forceinline__ v16h load_bT_frag(const _Float16* __restrict__ W, int ldw,
                                             int n0, int k0) {
  const int lane = threadIdx.x & 31;
  const int n  = n0 + (lane & 15);
  const int kb = k0 + ((lane >> 4) << 4);
  const _Float16* p = W + (size_t)n * ldw + kb;
  v16h b;
  *(v8h*)&b       = *(const v8h*)(p);
  *((v8h*)&b + 1) = *(const v8h*)(p + 8);
  return b;
}

// ---------------- Tensor Data Mover: DMA a rows x bk f16 tile (row stride ldw) to LDS ----
// D# built per CDNA5 ISA (group0: count/lds/global/type,
// group1: data_size=2B, tensor_dim0=ldw, tensor_dim1=rows, tile=bk x rows,
// tensor_dim0_stride=ldw). Groups 2/3 zero (2-D tensor). Tracked by TENSORcnt.
// This toolchain exposes the 6-arg builtin: (g0:u32x4, g1:i32x8, g2:i32x4, g3:i32x4,
// g4:i32x8, cpol:i32).
__device__ __forceinline__ void tdm_load_tile(const _Float16* gsrc, unsigned lds_off,
                                              int rows, int bk, int ldw) {
  const unsigned long long ga = (unsigned long long)(uintptr_t)gsrc;
  u32x4 g0;
  g0[0] = 1u;                                              // count=1 (valid user D#)
  g0[1] = lds_off;                                         // lds_addr (bytes)
  g0[2] = (unsigned)(ga & 0xFFFFFFFFu);                    // global_addr[31:0]
  g0[3] = (unsigned)((ga >> 32) & 0x01FFFFFFu) | (2u << 30); // addr[56:32] | type=2
  const unsigned td0 = (unsigned)ldw;                      // tensor_dim0 (elements)
  const unsigned td1 = (unsigned)rows;                     // tensor_dim1
  const unsigned long long st0 = (unsigned long long)ldw;  // tensor_dim0_stride
  i32x8 g1;
  g1[0] = (int)(1u << 16);                                 // data_size=1 -> 2 bytes
  g1[1] = (int)((td0 & 0xFFFFu) << 16);                    // tensor_dim0[15:0]
  g1[2] = (int)(((td0 >> 16) & 0xFFFFu) | ((td1 & 0xFFFFu) << 16));
  g1[3] = (int)(((td1 >> 16) & 0xFFFFu) | (((unsigned)bk & 0xFFFFu) << 16)); // tile_dim0
  g1[4] = (int)((unsigned)rows & 0xFFFFu);                 // tile_dim1
  g1[5] = (int)(st0 & 0xFFFFFFFFu);                        // dim0_stride[31:0]
  g1[6] = (int)((st0 >> 32) & 0xFFFFu);                    // dim0_stride[47:32]
  g1[7] = 0;
  const i32x4 z4 = {0, 0, 0, 0};
  const i32x8 z8 = {0, 0, 0, 0, 0, 0, 0, 0};
  __builtin_amdgcn_tensor_load_to_lds(g0, g1, z4, z4, z8, 0);
}

__device__ __forceinline__ unsigned lds_offset_of(const void* p) {
  return (unsigned)(uintptr_t)p; // flat LDS aperture: addr[31:0] is the LDS offset
}

// ---------------- GEMM: out = act(A @ W^T + bias) ----------------
// A: MxK f16 row-major, W: NxK f16 row-major. Block 256 thr = 8 waves, tile 128x64.
// Weight tile (64 x BK) is staged into LDS by the Tensor Data Mover, double buffered
// so DMA for chunk kc+1 overlaps WMMA on chunk kc. act: 0 = none, 1 = exact GELU.
__global__ void gemm_wmma_kernel(const _Float16* __restrict__ A,
                                 const _Float16* __restrict__ W,
                                 const float* __restrict__ bias,
                                 float* __restrict__ out32,
                                 _Float16* __restrict__ out16,
                                 int M, int N, int K, int act) {
  __shared__ __attribute__((aligned(16))) _Float16 s_w[2][64 * BK_];
  const int lane = threadIdx.x & 31;
  const int wave = threadIdx.x >> 5;
  const int row0 = blockIdx.x * 128 + wave * 16;
  const int n0b  = blockIdx.y * 64;
  const bool issuer = (wave == 0);

  const _Float16* wbase = W + (size_t)n0b * K;
  const int nk = K / BK_;

  if (issuer) tdm_load_tile(wbase, lds_offset_of(&s_w[0][0]), 64, BK_, K);

  v8f acc[4] = {};
  for (int kc = 0; kc < nk; ++kc) {
    if (kc + 1 < nk) {
      if (issuer) {
        tdm_load_tile(wbase + (size_t)(kc + 1) * BK_,
                      lds_offset_of(&s_w[(kc + 1) & 1][0]), 64, BK_, K);
        __builtin_amdgcn_s_wait_tensorcnt(1); // chunk kc landed; kc+1 in flight
      }
      // prefetch next A chunk (one byte per lane pulls the cachelines)
      __builtin_prefetch(A + (size_t)(row0 + (lane & 15)) * K + (size_t)(kc + 1) * BK_ +
                             ((lane >> 4) << 3), 0, 0);
    } else if (issuer) {
      __builtin_amdgcn_s_wait_tensorcnt(0);
    }
    __syncthreads();

    const _Float16* wt = &s_w[kc & 1][0];
#pragma unroll
    for (int k0 = 0; k0 < BK_; k0 += 32) {
      v16h a = load_a_frag(A, K, row0, kc * BK_ + k0);
#pragma unroll
      for (int nt = 0; nt < 4; ++nt) {
        v16h b = load_bT_frag(wt, BK_, nt * 16, k0); // B frags from LDS (shared by 8 waves)
        acc[nt] = __builtin_amdgcn_wmma_f32_16x16x32_f16(false, a, false, b,
                                                         (short)0, acc[nt], false, false);
      }
    }
    __syncthreads(); // everyone done reading buf[kc&1] before it is re-filled
  }

  // epilogue: C layout lane n = n0+(lane&15), VGPR r -> M = row0 + 8*hi + r
  const int n_in = lane & 15;
  const int mb   = row0 + ((lane >> 4) << 3);
#pragma unroll
  for (int nt = 0; nt < 4; ++nt) {
    const int n = n0b + nt * 16 + n_in;
    const float bn = bias ? bias[n] : 0.0f;
#pragma unroll
    for (int r = 0; r < 8; ++r) {
      float x = acc[nt][r] + bn;
      if (act == 1) x = 0.5f * x * (1.0f + erff(x * 0.70710678118654752f));
      const size_t idx = (size_t)(mb + r) * N + n;
      if (out32) out32[idx] = x;
      if (out16) out16[idx] = (_Float16)x;
    }
  }
}

// ---------------- f32 -> f16 convert ----------------
__global__ void cvt_kernel(const float* __restrict__ src, _Float16* __restrict__ dst,
                           size_t n) {
  size_t i = (size_t)blockIdx.x * 256 + threadIdx.x;
  if (i < n) dst[i] = (_Float16)src[i];
}

// ---------------- embedding: conv1d(circular,k=3) + sinusoid PE + temporal ----------------
__global__ void embed_kernel(const float* __restrict__ x_enc, const float* __restrict__ x_dec,
                             const float* __restrict__ xm_enc, const float* __restrict__ xm_dec,
                             const float* __restrict__ conv_w, const float* __restrict__ temporal_w,
                             float* __restrict__ h32, _Float16* __restrict__ h16) {
  size_t gid = (size_t)blockIdx.x * 256 + threadIdx.x; // NTOK_*D_
  const int d = (int)(gid % D_);
  const size_t rem = gid / D_;
  const int t = (int)(rem % T_);
  const int b = (int)(rem / T_);

  float acc = 0.0f;
#pragma unroll
  for (int k = 0; k < 3; ++k) {
    int s = t + k - 1;
    if (s < 0) s += T_;
    if (s >= T_) s -= T_;
    const float* xr = (s < SEQ_) ? (x_enc + ((size_t)b * SEQ_ + s) * CIN_)
                                 : (x_dec + ((size_t)b * PRED_ + (s - SEQ_)) * CIN_);
    const float* wk = conv_w + (size_t)d * (CIN_ * 3) + k; // conv_w[d][c][k]
#pragma unroll
    for (int c = 0; c < CIN_; ++c) acc += xr[c] * wk[c * 3];
  }
  // positional embedding
  const int i2 = d & ~1;
  const float div = expf(-logf(10000.0f) * (float)i2 / (float)D_);
  const float ang = (float)t * div;
  acc += (d & 1) ? cosf(ang) : sinf(ang);
  // temporal embedding
  const float* xmr = (t < SEQ_) ? (xm_enc + ((size_t)b * SEQ_ + t) * MARK_)
                                : (xm_dec + ((size_t)b * PRED_ + (t - SEQ_)) * MARK_);
  const float* tw = temporal_w + (size_t)d * MARK_;
#pragma unroll
  for (int m = 0; m < MARK_; ++m) acc += xmr[m] * tw[m];

  h32[gid] = acc;
  h16[gid] = (_Float16)acc;
}

// ---------------- LSH hashing: rotations, argmax bucket, packed sort key ----------------
__global__ void hash_kernel(const float* __restrict__ qk, const float* __restrict__ rot,
                            unsigned long long* __restrict__ keys) {
  const int idx = blockIdx.x * 256 + threadIdx.x; // BH_ * T_
  const int bh = idx / T_;
  const int t  = idx % T_;
  const int b  = bh / H_;
  const int hh = bh % H_;
  const float* q = qk + ((size_t)b * T_ + t) * D_ + hh * DH_;
  float qr[DH_];
#pragma unroll
  for (int f = 0; f < DH_; ++f) qr[f] = q[f];

  for (int r = 0; r < NHASH_; ++r) {
    float dots[32];
#pragma unroll 4
    for (int i = 0; i < 32; ++i) {
      float s = 0.0f;
      const float* rp = rot + r * 32 + i; // rot[f][r][i], stride over f = NHASH_*32
#pragma unroll
      for (int f = 0; f < DH_; ++f) s += qr[f] * rp[(size_t)f * (NHASH_ * 32)];
      dots[i] = s;
    }
    float best = -3.4e38f; int bi = 0;
    for (int i = 0; i < 64; ++i) {
      float v = (i < 32) ? dots[i] : -dots[i - 32];
      if (v > best) { best = v; bi = i; }
    }
    const int bucket = bi + r * NB_;                        // + hash-round offset
    const unsigned long long bt = (unsigned long long)(bucket * T_ + t);
    const unsigned long long ticker = (unsigned long long)(r * T_ + t);
    keys[(size_t)bh * NS_ + r * T_ + t] = (bt << 13) | ticker; // stable sort key
  }
}

// ---------------- bitonic sort of 8192 packed keys per bh row (LDS resident) ----------------
__global__ void sort_kernel(const unsigned long long* __restrict__ keys,
                            int* __restrict__ sidx, int* __restrict__ undo) {
  __shared__ unsigned long long s[NS_]; // 64 KB
  const int bh = blockIdx.x;
  const int tid = threadIdx.x;
  for (int i = tid; i < NS_; i += 256) s[i] = keys[(size_t)bh * NS_ + i];
  __syncthreads();
  for (int k = 2; k <= NS_; k <<= 1) {
    for (int j = k >> 1; j > 0; j >>= 1) {
      for (int i = tid; i < NS_; i += 256) {
        const int ixj = i ^ j;
        if (ixj > i) {
          const unsigned long long a = s[i], c = s[ixj];
          const bool up = ((i & k) == 0);
          if ((a > c) == up) { s[i] = c; s[ixj] = a; }
        }
      }
      __syncthreads();
    }
  }
  for (int i = tid; i < NS_; i += 256) {
    const int sv = (int)(s[i] & 8191ULL);
    sidx[(size_t)bh * NS_ + i] = sv;
    undo[(size_t)bh * NS_ + sv] = i;
  }
}

// ---------------- gather into sorted order: bq (f16), bk (normalized f16), bv (f16) ----------------
__global__ void gather_kernel(const float* __restrict__ qk, const float* __restrict__ v,
                              const int* __restrict__ sidx,
                              _Float16* __restrict__ bq, _Float16* __restrict__ bk,
                              _Float16* __restrict__ bv, int* __restrict__ st) {
  const size_t gid = (size_t)blockIdx.x * 256 + threadIdx.x; // BH_ * NS_
  const int bh = (int)(gid / NS_);
  const int b  = bh / H_;
  const int hh = bh % H_;
  const int j  = sidx[gid];
  const int t  = j % T_;
  st[gid] = t;
  const float* qrow = qk + ((size_t)b * T_ + t) * D_ + hh * DH_;
  const float* vrow = v  + ((size_t)b * T_ + t) * D_ + hh * DH_;
  float qv[DH_];
  float nrm = 0.0f;
#pragma unroll
  for (int f = 0; f < DH_; ++f) { qv[f] = qrow[f]; nrm += qv[f] * qv[f]; }
  nrm = fmaxf(sqrtf(nrm), 1e-12f);
  const float inv = 1.0f / nrm;
  _Float16* bqp = bq + gid * DH_;
  _Float16* bkp = bk + gid * DH_;
  _Float16* bvp = bv + gid * DH_;
#pragma unroll
  for (int f = 0; f < DH_; ++f) {
    bqp[f] = (_Float16)qv[f];
    bkp[f] = (_Float16)(qv[f] * inv);
    bvp[f] = (_Float16)vrow[f];
  }
}

// ---------------- chunked LSH attention: one wave32 per (bh, chunk) ----------------
__global__ void __launch_bounds__(32)
attn_chunk_kernel(const _Float16* __restrict__ bq, const _Float16* __restrict__ bk,
                  const _Float16* __restrict__ bv, const int* __restrict__ st,
                  float* __restrict__ bo, float* __restrict__ slog) {
  __shared__ int      s_qt[32];
  __shared__ int      s_kt[64];
  __shared__ __attribute__((aligned(16))) float    s_dots[32 * 64];
  __shared__ __attribute__((aligned(16))) _Float16 s_p[32 * 64];

  const int bh   = blockIdx.x / NCHUNK_;
  const int c    = blockIdx.x % NCHUNK_;
  const int prev = (c + NCHUNK_ - 1) % NCHUNK_; // look-one-back (rolled over chunk axis)
  const int lane = threadIdx.x;
  const size_t base = (size_t)bh * NS_;

  s_qt[lane]      = st[base + c * BUCKET_ + lane];
  s_kt[lane]      = st[base + c * BUCKET_ + lane];
  s_kt[32 + lane] = st[base + prev * BUCKET_ + lane];
  __syncthreads();

  // ---- phase 1: dots = bq @ bk^T ----
  v8f acc[2][4] = {};
  for (int k0 = 0; k0 < DH_; k0 += 32) {
    v16h a[2];
#pragma unroll
    for (int mt = 0; mt < 2; ++mt)
      a[mt] = load_a_frag(bq + base * DH_, DH_, c * BUCKET_ + mt * 16, k0);
#pragma unroll
    for (int nt = 0; nt < 4; ++nt) {
      const int n = nt * 16 + (lane & 15);
      const int srow = (n < 32) ? (c * BUCKET_ + n) : (prev * BUCKET_ + (n - 32));
      const _Float16* p = bk + (base + srow) * DH_ + k0 + ((lane >> 4) << 4);
      v16h b;
      *(v8h*)&b       = *(const v8h*)(p);
      *((v8h*)&b + 1) = *(const v8h*)(p + 8);
#pragma unroll
      for (int mt = 0; mt < 2; ++mt)
        acc[mt][nt] = __builtin_amdgcn_wmma_f32_16x16x32_f16(false, a[mt], false, b,
                                                             (short)0, acc[mt][nt],
                                                             false, false);
    }
  }

  // scale + self-token mask, spill 32x64 tile to LDS
  const int n_in = lane & 15;
  const int hi8  = (lane >> 4) << 3;
#pragma unroll
  for (int mt = 0; mt < 2; ++mt) {
#pragma unroll
    for (int nt = 0; nt < 4; ++nt) {
      const int j = nt * 16 + n_in;
      const int ktj = s_kt[j];
#pragma unroll
      for (int r = 0; r < 8; ++r) {
        const int m = mt * 16 + hi8 + r;
        float vdt = acc[mt][nt][r] * 0.125f; // DH^-0.5
        if (s_qt[m] == ktj) vdt = SELF_ATTN_VALUE;
        s_dots[m * 64 + j] = vdt;
      }
    }
  }
  __syncthreads();

  // ---- row softmax (lane == row) ----
  {
    float mx = -3.4e38f;
    for (int j = 0; j < 64; ++j) mx = fmaxf(mx, s_dots[lane * 64 + j]);
    float ssum = 0.0f;
    for (int j = 0; j < 64; ++j) ssum += expf(s_dots[lane * 64 + j] - mx);
    const float lse = mx + logf(ssum);
    for (int j = 0; j < 64; ++j)
      s_p[lane * 64 + j] = (_Float16)expf(s_dots[lane * 64 + j] - lse);
    slog[base + c * BUCKET_ + lane] = lse;
  }
  __syncthreads();

  // ---- phase 2: bo = p @ bv ----
  v8f acc2[2][4] = {};
  for (int j0 = 0; j0 < 64; j0 += 32) {
    v16h a[2];
#pragma unroll
    for (int mt = 0; mt < 2; ++mt)
      a[mt] = load_a_frag((const _Float16*)s_p, 64, mt * 16, j0);
#pragma unroll
    for (int nt = 0; nt < 4; ++nt) {
      const int n = nt * 16 + n_in;               // dh column
      const int kb = j0 + ((lane >> 4) << 4);
      v16h b;
#pragma unroll
      for (int e = 0; e < 16; ++e) {
        const int j = kb + e;
        const int srow = (j < 32) ? (c * BUCKET_ + j) : (prev * BUCKET_ + (j - 32));
        b[e] = bv[(base + srow) * DH_ + n];
      }
#pragma unroll
      for (int mt = 0; mt < 2; ++mt)
        acc2[mt][nt] = __builtin_amdgcn_wmma_f32_16x16x32_f16(false, a[mt], false, b,
                                                              (short)0, acc2[mt][nt],
                                                              false, false);
    }
  }
#pragma unroll
  for (int mt = 0; mt < 2; ++mt) {
#pragma unroll
    for (int nt = 0; nt < 4; ++nt) {
#pragma unroll
      for (int r = 0; r < 8; ++r) {
        const int m = mt * 16 + hi8 + r;
        bo[(base + c * BUCKET_ + m) * DH_ + nt * 16 + n_in] = acc2[mt][nt][r];
      }
    }
  }
}

// ---------------- unsort + multi-round logsumexp combine, unmerge heads ----------------
__global__ void combine_kernel(const float* __restrict__ bo, const float* __restrict__ slog,
                               const int* __restrict__ undo, _Float16* __restrict__ attn16) {
  const size_t gid = (size_t)blockIdx.x * 256 + threadIdx.x; // BH_*T_*DH_
  const int d = (int)(gid % DH_);
  const size_t rem = gid / DH_;
  const int t  = (int)(rem % T_);
  const int bh = (int)(rem / T_);
  const int b  = bh / H_;
  const int hh = bh % H_;

  float lg[NHASH_], ov[NHASH_];
  float mx = -3.4e38f;
#pragma unroll
  for (int r = 0; r < NHASH_; ++r) {
    const int j = undo[(size_t)bh * NS_ + r * T_ + t];
    lg[r] = slog[(size_t)bh * NS_ + j];
    ov[r] = bo[((size_t)bh * NS_ + j) * DH_ + d];
    mx = fmaxf(mx, lg[r]);
  }
  float ssum = 0.0f;
#pragma unroll
  for (int r = 0; r < NHASH_; ++r) ssum += expf(lg[r] - mx);
  const float lse = mx + logf(ssum);
  float o = 0.0f;
#pragma unroll
  for (int r = 0; r < NHASH_; ++r) o += expf(lg[r] - lse) * ov[r];
  attn16[((size_t)b * T_ + t) * D_ + hh * DH_ + d] = (_Float16)o;
}

// ---------------- residual + layernorm (row per block, D=512, 2 elems/thread) ----------------
__global__ void ln_kernel(const float* __restrict__ x, const float* __restrict__ addv,
                          const float* __restrict__ g, const float* __restrict__ bb,
                          float* __restrict__ out32, _Float16* __restrict__ out16) {
  __shared__ float r1[256], r2[256];
  const size_t base = (size_t)blockIdx.x * D_;
  const int tid = threadIdx.x;
  float v0 = x[base + tid];
  float v1 = x[base + tid + 256];
  if (addv) { v0 += addv[base + tid]; v1 += addv[base + tid + 256]; }
  r1[tid] = v0 + v1;
  r2[tid] = v0 * v0 + v1 * v1;
  __syncthreads();
  for (int s2 = 128; s2 > 0; s2 >>= 1) {
    if (tid < s2) { r1[tid] += r1[tid + s2]; r2[tid] += r2[tid + s2]; }
    __syncthreads();
  }
  const float mean = r1[0] / (float)D_;
  const float var  = r2[0] / (float)D_ - mean * mean;
  const float inv  = rsqrtf(var + 1e-5f);
  const float o0 = (v0 - mean) * inv * g[tid] + bb[tid];
  const float o1 = (v1 - mean) * inv * g[tid + 256] + bb[tid + 256];
  out32[base + tid] = o0;
  out32[base + tid + 256] = o1;
  if (out16) {
    out16[base + tid] = (_Float16)o0;
    out16[base + tid + 256] = (_Float16)o1;
  }
}

// ---------------- final projection D -> C_IN on the last PRED tokens ----------------
__global__ void proj_kernel(const float* __restrict__ hf, const float* __restrict__ pw,
                            const float* __restrict__ pb, float* __restrict__ out) {
  const int gid = blockIdx.x * 256 + threadIdx.x;
  if (gid >= B_ * PRED_ * CIN_) return;
  const int cc = gid % CIN_;
  const int p  = (gid / CIN_) % PRED_;
  const int b  = gid / (CIN_ * PRED_);
  const float* hr = hf + ((size_t)b * T_ + SEQ_ + p) * D_;
  const float* wr = pw + (size_t)cc * D_;
  float s = pb[cc];
#pragma unroll 8
  for (int f = 0; f < D_; ++f) s += hr[f] * wr[f];
  out[gid] = s;
}

// ---------------- host orchestration ----------------
extern "C" void kernel_launch(void* const* d_in, const int* in_sizes, int n_in,
                              void* d_out, int out_size, void* d_ws, size_t ws_size,
                              hipStream_t stream) {
  (void)in_sizes; (void)n_in; (void)out_size; (void)ws_size;
  const float* x_enc      = (const float*)d_in[0];
  const float* xm_enc     = (const float*)d_in[1];
  const float* x_dec      = (const float*)d_in[2];
  const float* xm_dec     = (const float*)d_in[3];
  const float* conv_w     = (const float*)d_in[4];
  const float* temporal_w = (const float*)d_in[5];
  const float* Wqk        = (const float*)d_in[6];
  const float* Wv         = (const float*)d_in[7];
  const float* Wo_w       = (const float*)d_in[8];
  const float* Wo_b       = (const float*)d_in[9];
  const float* rot        = (const float*)d_in[10];
  const float* ln1_g      = (const float*)d_in[11];
  const float* ln1_b      = (const float*)d_in[12];
  const float* ff1_w      = (const float*)d_in[13];
  const float* ff1_b      = (const float*)d_in[14];
  const float* ff2_w      = (const float*)d_in[15];
  const float* ff2_b      = (const float*)d_in[16];
  const float* ln2_g      = (const float*)d_in[17];
  const float* ln2_b      = (const float*)d_in[18];
  const float* lnf_g      = (const float*)d_in[19];
  const float* lnf_b      = (const float*)d_in[20];
  const float* proj_w     = (const float*)d_in[21];
  const float* proj_b     = (const float*)d_in[22];

  // bump allocator over workspace
  char* wp = (char*)d_ws;
  auto alloc = [&](size_t bytes) -> void* {
    void* r = (void*)wp;
    wp += (bytes + 255) & ~(size_t)255;
    return r;
  };
  float*    h32    = (float*)alloc((size_t)NTOK_ * D_ * sizeof(float));
  _Float16* h16    = (_Float16*)alloc((size_t)NTOK_ * D_ * sizeof(_Float16));
  float*    qk32   = (float*)alloc((size_t)NTOK_ * D_ * sizeof(float));
  float*    v32    = (float*)alloc((size_t)NTOK_ * D_ * sizeof(float));
  float*    tmp32  = (float*)alloc((size_t)NTOK_ * D_ * sizeof(float));
  _Float16* attn16 = (_Float16*)alloc((size_t)NTOK_ * D_ * sizeof(_Float16));
  _Float16* w16    = (_Float16*)alloc((size_t)DFF_ * D_ * sizeof(_Float16));
  _Float16* y16    = (_Float16*)alloc((size_t)NTOK_ * DFF_ * sizeof(_Float16));
  unsigned long long* keys = (unsigned long long*)alloc((size_t)BH_ * NS_ * 8);
  int*      sidx   = (int*)alloc((size_t)BH_ * NS_ * sizeof(int));
  int*      undo   = (int*)alloc((size_t)BH_ * NS_ * sizeof(int));
  int*      starr  = (int*)alloc((size_t)BH_ * NS_ * sizeof(int));
  _Float16* bq16   = (_Float16*)alloc((size_t)BH_ * NS_ * DH_ * sizeof(_Float16));
  _Float16* bk16   = (_Float16*)alloc((size_t)BH_ * NS_ * DH_ * sizeof(_Float16));
  _Float16* bv16   = (_Float16*)alloc((size_t)BH_ * NS_ * DH_ * sizeof(_Float16));
  float*    bo32   = (float*)alloc((size_t)BH_ * NS_ * DH_ * sizeof(float));
  float*    slog   = (float*)alloc((size_t)BH_ * NS_ * sizeof(float));

  auto cvt = [&](const float* src, _Float16* dst, size_t n) {
    cvt_kernel<<<dim3((unsigned)((n + 255) / 256)), 256, 0, stream>>>(src, dst, n);
  };
  auto gemm = [&](const _Float16* Aa, const _Float16* Ww, const float* bias,
                  float* o32, _Float16* o16, int M, int N, int K, int act) {
    dim3 g(M / 128, N / 64);
    gemm_wmma_kernel<<<g, 256, 0, stream>>>(Aa, Ww, bias, o32, o16, M, N, K, act);
  };

  // embedding
  embed_kernel<<<NTOK_ * D_ / 256, 256, 0, stream>>>(x_enc, x_dec, xm_enc, xm_dec,
                                                     conv_w, temporal_w, h32, h16);

  for (int l = 0; l < L_; ++l) {
    const size_t wDD = (size_t)l * D_ * D_;
    // shared-QK and V projections
    cvt(Wqk + wDD, w16, (size_t)D_ * D_);
    gemm(h16, w16, nullptr, qk32, nullptr, NTOK_, D_, D_, 0);
    cvt(Wv + wDD, w16, (size_t)D_ * D_);
    gemm(h16, w16, nullptr, v32, nullptr, NTOK_, D_, D_, 0);
    // LSH: hash -> stable sort -> gather
    hash_kernel<<<BH_ * T_ / 256, 256, 0, stream>>>(
        qk32, rot + (size_t)l * DH_ * NHASH_ * (NB_ / 2), keys);
    sort_kernel<<<BH_, 256, 0, stream>>>(keys, sidx, undo);
    gather_kernel<<<BH_ * NS_ / 256, 256, 0, stream>>>(qk32, v32, sidx,
                                                       bq16, bk16, bv16, starr);
    // chunked attention (WMMA) + multi-round combine
    attn_chunk_kernel<<<BH_ * NCHUNK_, 32, 0, stream>>>(bq16, bk16, bv16, starr,
                                                        bo32, slog);
    combine_kernel<<<BH_ * T_ * DH_ / 256, 256, 0, stream>>>(bo32, slog, undo, attn16);
    // output projection + residual LN
    cvt(Wo_w + wDD, w16, (size_t)D_ * D_);
    gemm(attn16, w16, Wo_b + (size_t)l * D_, tmp32, nullptr, NTOK_, D_, D_, 0);
    ln_kernel<<<NTOK_, 256, 0, stream>>>(h32, tmp32, ln1_g + (size_t)l * D_,
                                         ln1_b + (size_t)l * D_, h32, h16);
    // FFN: GELU(h@ff1^T+b1)@ff2^T+b2, residual LN
    cvt(ff1_w + (size_t)l * DFF_ * D_, w16, (size_t)DFF_ * D_);
    gemm(h16, w16, ff1_b + (size_t)l * DFF_, nullptr, y16, NTOK_, DFF_, D_, 1);
    cvt(ff2_w + (size_t)l * D_ * DFF_, w16, (size_t)D_ * DFF_);
    gemm(y16, w16, ff2_b + (size_t)l * D_, tmp32, nullptr, NTOK_, D_, DFF_, 0);
    ln_kernel<<<NTOK_, 256, 0, stream>>>(h32, tmp32, ln2_g + (size_t)l * D_,
                                         ln2_b + (size_t)l * D_, h32, h16);
  }

  // final LN + projection of last PRED tokens
  ln_kernel<<<NTOK_, 256, 0, stream>>>(h32, nullptr, lnf_g, lnf_b, tmp32, nullptr);
  proj_kernel<<<(B_ * PRED_ * CIN_ + 255) / 256, 256, 0, stream>>>(tmp32, proj_w, proj_b,
                                                                   (float*)d_out);
}
